// MoABlock_36404142801000
// MI455X (gfx1250) — matrix operand order repaired and checked by reading, not compiled, not run on hardware
//
#include <hip/hip_runtime.h>
#include <hip/hip_bf16.h>

// ---------------------------------------------------------------------------
// MoA block for MI455X (gfx1250, wave32). All GEMM-shaped math via
// v_wmma_f32_16x16x32_f16 (f16 operands, f32 accumulate). ~82 GFLOP vs ~50 MB
// traffic -> matrix-unit bound. This revision:
//  - Weights pre-converted to f16 once, so BOTH GEMM operand tiles stream via
//    gfx1250 async VMEM->LDS (global_load_async_to_lds_b128).
//  - Double-buffered LDS tiles: async copy of tile t+1 overlaps the 4 WMMAs of
//    tile t; single s_wait_asynccnt + one barrier per K-step.
//  - Power-of-two L: shift/mask instead of integer division in epilogues.
// ---------------------------------------------------------------------------

typedef __attribute__((ext_vector_type(16))) _Float16 v16h;
typedef __attribute__((ext_vector_type(8)))  _Float16 v8h;
typedef __attribute__((ext_vector_type(8)))  float    v8f;

#define EPS 1e-5f

__device__ __forceinline__ v8f zero_v8f() {
  v8f z;
#pragma unroll
  for (int i = 0; i < 8; ++i) z[i] = 0.0f;
  return z;
}

__device__ __forceinline__ v8f wmma_f16(v16h a, v16h b, v8f c) {
  return __builtin_amdgcn_wmma_f32_16x16x32_f16(
      false, a, false, b, (short)0, c, false, false);
}

// Async VMEM -> LDS copy of 16 bytes per lane (gfx1250 GLOBAL_LOAD_ASYNC).
__device__ __forceinline__ void async_copy_b128(unsigned dst_lds, const void* src) {
  asm volatile("global_load_async_to_lds_b128 %0, %1, off"
               :: "v"(dst_lds), "v"(src) : "memory");
}
__device__ __forceinline__ void async_wait() {
  asm volatile("s_wait_asynccnt 0x0" ::: "memory");
}
__device__ __forceinline__ unsigned lds_off(const void* p) {
  return (unsigned)(size_t)p;  // LDS aperture keeps DS offset in low 32 bits
}

// A fragment 16x32 (MxK), row-major source: elem(m,k) = p[m*ld + k]
// lanes 0-15: M=lane, K={0..7,16..23}; lanes 16-31: K+=8. 2x16B/lane -> b128.
__device__ __forceinline__ v16h frag_A_rm(const _Float16* p, int ld) {
  int l = threadIdx.x & 31;
  int m = l & 15;
  int kb = (l < 16) ? 0 : 8;
  v16h f;
#pragma unroll
  for (int i = 0; i < 8; ++i) {
    int k = kb + ((i < 4) ? (2 * i) : (16 + 2 * (i - 4)));
    f[2 * i]     = p[m * ld + k];
    f[2 * i + 1] = p[m * ld + k + 1];
  }
  return f;
}

// B fragment 32x16 (KxN), N-major source: elem(k,n) = p[n*ld + k]
// lanes 0-15: N=lane, K={0..7,16..23}; lanes 16-31: N=lane-16, K+=8. 2x16B/lane.
__device__ __forceinline__ v16h frag_B_tr(const _Float16* p, int ld) {
  int l = threadIdx.x & 31;
  int n = l & 15;
  int kb = (l < 16) ? 0 : 8;
  v16h f;
#pragma unroll
  for (int i = 0; i < 8; ++i) {
    int k = kb + ((i < 4) ? (2 * i) : (16 + 2 * (i - 4)));
    f[2 * i]     = p[n * ld + k];
    f[2 * i + 1] = p[n * ld + k + 1];
  }
  return f;
}

// ---------------------------------------------------------------------------
// K0: one-shot weight convert fp32 -> f16 (8 elems/thread, b128 in/out)
// ---------------------------------------------------------------------------
__global__ __launch_bounds__(256) void wcvt_kernel(
    const float* __restrict__ src, _Float16* __restrict__ dst, int n8) {
  int i = blockIdx.x * 256 + threadIdx.x;
  if (i < n8) {
    const float* s = src + (size_t)i * 8;
    float4 d0 = *(const float4*)s;
    float4 d1 = *(const float4*)(s + 4);
    v8h hv;
    hv[0] = (_Float16)d0.x; hv[1] = (_Float16)d0.y;
    hv[2] = (_Float16)d0.z; hv[3] = (_Float16)d0.w;
    hv[4] = (_Float16)d1.x; hv[5] = (_Float16)d1.y;
    hv[6] = (_Float16)d1.z; hv[7] = (_Float16)d1.w;
    *(v8h*)(dst + (size_t)i * 8) = hv;
  }
}

// ---------------------------------------------------------------------------
// K1: per-row LayerNorm, fp32 in -> f16 out. grid.x = B*L rows, 256 threads.
// ---------------------------------------------------------------------------
__global__ __launch_bounds__(256) void ln_kernel(
    const float* __restrict__ x, const float* __restrict__ gw,
    const float* __restrict__ gb, _Float16* __restrict__ xn) {
  __shared__ float r1[256], r2[256];
  __shared__ float mb, rb;
  int tid = threadIdx.x;
  const float* xr = x + (size_t)blockIdx.x * 512;
  float v0 = xr[tid], v1 = xr[tid + 256];
  r1[tid] = v0 + v1;
  r2[tid] = v0 * v0 + v1 * v1;
  __syncthreads();
#pragma unroll
  for (int o = 128; o > 0; o >>= 1) {
    if (tid < o) { r1[tid] += r1[tid + o]; r2[tid] += r2[tid + o]; }
    __syncthreads();
  }
  if (tid == 0) {
    float m = r1[0] * (1.0f / 512.0f);
    float var = r2[0] * (1.0f / 512.0f) - m * m;
    mb = m; rb = rsqrtf(var + EPS);
  }
  __syncthreads();
  float m = mb, r = rb;
  _Float16* dst = xn + (size_t)blockIdx.x * 512;
  dst[tid]       = (_Float16)((v0 - m) * r * gw[tid] + gb[tid]);
  dst[tid + 256] = (_Float16)((v1 - m) * r * gw[tid + 256] + gb[tid + 256]);
}

// ---------------------------------------------------------------------------
// K2: pooled = mean over L of xn.  grid.x = B, 256 threads.
// ---------------------------------------------------------------------------
__global__ __launch_bounds__(256) void pool_kernel(
    const _Float16* __restrict__ xn, float* __restrict__ pooled, int L) {
  int b = blockIdx.x, tid = threadIdx.x;
  float inv = 1.0f / (float)L;
#pragma unroll
  for (int c = 0; c < 2; ++c) {
    int d = tid + c * 256;
    float s = 0.0f;
    for (int l = 0; l < L; ++l) s += (float)xn[((size_t)b * L + l) * 512 + d];
    pooled[b * 512 + d] = s * inv;
  }
}

// ---------------------------------------------------------------------------
// K3: router path (tiny VALU): r = pooled + s*LN(emb); MLP(LN(r)); gates.
// ---------------------------------------------------------------------------
__device__ __forceinline__ float block_sum(float v, float* red) {
  int tid = threadIdx.x;
  red[tid] = v; __syncthreads();
#pragma unroll
  for (int o = 128; o > 0; o >>= 1) {
    if (tid < o) red[tid] += red[tid + o];
    __syncthreads();
  }
  float r = red[0]; __syncthreads();
  return r;
}

__global__ __launch_bounds__(256) void router_kernel(
    const float* __restrict__ pooled, const float* __restrict__ emb,
    const float* __restrict__ eg, const float* __restrict__ eb,
    const float* __restrict__ gscale,
    const float* __restrict__ mg, const float* __restrict__ mbv,
    const float* __restrict__ W1, const float* __restrict__ b1,
    const float* __restrict__ W2, const float* __restrict__ b2,
    const float* __restrict__ Wg, const float* __restrict__ bg,
    float* __restrict__ latents, float* __restrict__ gates) {
  __shared__ float red[256];
  __shared__ float geln[512], hln[512], h1[512], h2[512];
  int tid = threadIdx.x;
  float scale = gscale[0];
  for (int g = 0; g < 3; ++g) {
    float e0 = emb[g * 512 + tid], e1 = emb[g * 512 + 256 + tid];
    float s  = block_sum(e0 + e1, red);
    float s2 = block_sum(e0 * e0 + e1 * e1, red);
    float m = s * (1.0f / 512.0f);
    float r = rsqrtf(s2 * (1.0f / 512.0f) - m * m + EPS);
    geln[tid]       = (e0 - m) * r * eg[tid] + eb[tid];
    geln[tid + 256] = (e1 - m) * r * eg[tid + 256] + eb[tid + 256];
    __syncthreads();
    for (int b = 0; b < 8; ++b) {
      const float* pr = pooled + (size_t)(g * 8 + b) * 512;
      float r0 = pr[tid] + scale * geln[tid];
      float r1 = pr[tid + 256] + scale * geln[tid + 256];
      float ss  = block_sum(r0 + r1, red);
      float ss2 = block_sum(r0 * r0 + r1 * r1, red);
      float mm = ss * (1.0f / 512.0f);
      float rr = rsqrtf(ss2 * (1.0f / 512.0f) - mm * mm + EPS);
      hln[tid]       = (r0 - mm) * rr * mg[tid] + mbv[tid];
      hln[tid + 256] = (r1 - mm) * rr * mg[tid + 256] + mbv[tid + 256];
      __syncthreads();
#pragma unroll
      for (int jj = 0; jj < 2; ++jj) {
        int j = tid + jj * 256;
        const float* wr = W1 + (size_t)j * 512;
        float acc = b1[j];
        for (int d = 0; d < 512; ++d) acc += hln[d] * wr[d];
        h1[j] = 0.5f * acc * (1.0f + erff(acc * 0.70710678118f));  // exact GELU
      }
      __syncthreads();
#pragma unroll
      for (int jj = 0; jj < 2; ++jj) {
        int j = tid + jj * 256;
        const float* wr = W2 + (size_t)j * 512;
        float acc = b2[j];
        for (int d = 0; d < 512; ++d) acc += h1[d] * wr[d];
        h2[j] = acc;
        latents[((size_t)b * 3 + g) * 512 + j] = acc;
      }
      __syncthreads();
      if (tid == 0) {
        float lg[4], mx = -3.0e38f;
        for (int e = 0; e < 4; ++e) {
          float acc = bg[e];
          const float* wr = Wg + e * 512;
          for (int d = 0; d < 512; ++d) acc += h2[d] * wr[d];
          lg[e] = acc; mx = fmaxf(mx, acc);
        }
        float sm = 0.0f;
        for (int e = 0; e < 4; ++e) { lg[e] = __expf(lg[e] - mx); sm += lg[e]; }
        for (int e = 0; e < 4; ++e) gates[(b * 3 + g) * 4 + e] = lg[e] / sm;
      }
      __syncthreads();
    }
  }
}

// ---------------------------------------------------------------------------
// K4: grouped QKV projection GEMM via WMMA, double-buffered async staging.
// out[tok,n] = sum_d xn[tok,d] * Wh[n,d] + bias[n]
// q,k -> [b,e,h,l,dk];  v -> TRANSPOSED [b,e,h,dk,L].
// grid: x = 512/64 (N), y = Ntok/128 (M), z = 12 (e*3 + {q,k,v}).
// ---------------------------------------------------------------------------
__global__ __launch_bounds__(256) void proj_kernel(
    const _Float16* __restrict__ xn,
    const _Float16* __restrict__ Wqh, const float* __restrict__ bq,
    const _Float16* __restrict__ Wkh, const float* __restrict__ bk,
    const _Float16* __restrict__ Wvh, const float* __restrict__ bv,
    _Float16* __restrict__ qd, _Float16* __restrict__ kd,
    _Float16* __restrict__ vd, int L, int llog) {
  __shared__ alignas(16) _Float16 At[2][128][48];  // [m][k], 96B row stride
  __shared__ alignas(16) _Float16 Bs[2][64][40];   // [n][k], 80B row stride
  int e = blockIdx.z / 3, p = blockIdx.z % 3;
  const _Float16* Wh  = (p == 0 ? Wqh : (p == 1 ? Wkh : Wvh)) + (size_t)e * 512 * 512;
  const float* bias   = (p == 0 ? bq : (p == 1 ? bk : bv)) + e * 512;
  _Float16* dst = (p == 0 ? qd : (p == 1 ? kd : vd));
  int tid = threadIdx.x, wave = tid >> 5, lane = tid & 31;
  int wm = wave >> 1, wn = wave & 1;
  int mbase = blockIdx.y * 128, nbase = blockIdx.x * 64;

  auto stage = [&](int buf, int k0) {
#pragma unroll
    for (int c = 0; c < 2; ++c) {  // A: 128x32 halves
      int idx = tid + c * 256;
      int m = idx >> 2, kk = (idx & 3) * 8;
      async_copy_b128(lds_off(&At[buf][m][kk]),
                      xn + (size_t)(mbase + m) * 512 + k0 + kk);
    }
    {  // B: 64x32 halves (f16 weights, [n][k] tile)
      int n = tid >> 2, kk = (tid & 3) * 8;
      async_copy_b128(lds_off(&Bs[buf][n][kk]),
                      Wh + (size_t)(nbase + n) * 512 + k0 + kk);
    }
  };

  v8f acc[2][2];
#pragma unroll
  for (int i = 0; i < 2; ++i)
#pragma unroll
    for (int j = 0; j < 2; ++j) acc[i][j] = zero_v8f();

  stage(0, 0);
  async_wait();
  __syncthreads();
  for (int s = 0; s < 16; ++s) {       // K = 512 in steps of 32
    int cur = s & 1;
    if (s < 15) stage(cur ^ 1, (s + 1) * 32);   // overlap copy with WMMA
    v16h a0 = frag_A_rm(&At[cur][wm * 32][0], 48);
    v16h a1 = frag_A_rm(&At[cur][wm * 32 + 16][0], 48);
    v16h b0 = frag_B_tr(&Bs[cur][wn * 32][0], 40);
    v16h b1 = frag_B_tr(&Bs[cur][wn * 32 + 16][0], 40);
    acc[0][0] = wmma_f16(a0, b0, acc[0][0]);
    acc[0][1] = wmma_f16(a0, b1, acc[0][1]);
    acc[1][0] = wmma_f16(a1, b0, acc[1][0]);
    acc[1][1] = wmma_f16(a1, b1, acc[1][1]);
    if (s < 15) { async_wait(); __syncthreads(); }
  }
  // epilogue: +bias, scatter f16
  int mh = (lane >> 4) * 8;
#pragma unroll
  for (int i = 0; i < 2; ++i)
#pragma unroll
    for (int j = 0; j < 2; ++j) {
      int ncol = nbase + wn * 32 + j * 16 + (lane & 15);
      float bvv = bias[ncol];
      int h = ncol >> 6, kkf = ncol & 63;
#pragma unroll
      for (int r = 0; r < 8; ++r) {
        int m = mbase + wm * 32 + i * 16 + mh + r;   // token index
        int bidx = m >> llog, l = m & (L - 1);
        size_t off;
        if (p == 2)  // V transposed: [b,e,h,dk,L]
          off = ((((size_t)bidx * 4 + e) * 8 + h) * 64 + kkf) * (size_t)L + l;
        else         // Q,K: [b,e,h,l,dk]
          off = ((((size_t)bidx * 4 + e) * 8 + h) * (size_t)L + l) * 64 + kkf;
        dst[off] = (_Float16)(acc[i][j][r] + bvv);
      }
    }
}

// ---------------------------------------------------------------------------
// K5: attention per (b,e,h, 16-query-row tile).
// S = QK^T (WMMA), full S row in LDS fp32, shuffle softmax (unnormalized f16
// probs + rowsum), O = P V with V^T layout, K-dim split across wave halves.
// grid: x = L/16, y = B, z = E*H.  256 threads.
// ---------------------------------------------------------------------------
__global__ __launch_bounds__(256) void attn_kernel(
    const _Float16* __restrict__ q, const _Float16* __restrict__ k,
    const _Float16* __restrict__ vT, _Float16* __restrict__ o,
    int L, float scale) {
  __shared__ alignas(16) _Float16 qt[16][64];
  __shared__ float S[16][512];
  __shared__ alignas(16) _Float16 P[16][512];
  __shared__ float rowsum[16];
  __shared__ float redbuf[4][32][8];
  int b = blockIdx.y, e = blockIdx.z >> 3, h = blockIdx.z & 7;
  size_t head = (((size_t)b * 4 + e) * 8 + h) * (size_t)L * 64;
  const _Float16* qh = q + head + (size_t)blockIdx.x * 16 * 64;
  const _Float16* kh = k + head;
  const _Float16* vh = vT + head;   // interpreted [64][L]
  int tid = threadIdx.x, wave = tid >> 5, lane = tid & 31;

  {  // stage 16x64 q tile via async VMEM->LDS (even threads, 16B each)
    int idx = tid * 4;
    int m = idx >> 6, kk = idx & 63;
    if ((tid & 1) == 0)
      async_copy_b128(lds_off(&qt[m][kk]), qh + m * 64 + kk);
  }
  async_wait();
  __syncthreads();

  int nt_count = L >> 4;
  for (int nt = wave; nt < nt_count; nt += 8) {  // S tiles, waves cyclic
    v8f acc = zero_v8f();
    v16h a0 = frag_A_rm(&qt[0][0], 64);
    v16h b0 = frag_B_tr(kh + (size_t)nt * 16 * 64, 64);
    acc = wmma_f16(a0, b0, acc);
    v16h a1 = frag_A_rm(&qt[0][32], 64);
    v16h b1 = frag_B_tr(kh + (size_t)nt * 16 * 64 + 32, 64);
    acc = wmma_f16(a1, b1, acc);
    int col = nt * 16 + (lane & 15);
    int mh = (lane >> 4) * 8;
#pragma unroll
    for (int r = 0; r < 8; ++r) S[mh + r][col] = acc[r] * scale;
  }
  __syncthreads();

  {  // softmax: 16 threads per row, within-wave shuffle reductions
    int row = tid >> 4, sub = tid & 15;
    float m = -3.0e38f;
    for (int j = sub; j < L; j += 16) m = fmaxf(m, S[row][j]);
#pragma unroll
    for (int ofs = 8; ofs >= 1; ofs >>= 1) m = fmaxf(m, __shfl_xor(m, ofs, 32));
    float s = 0.0f;
    for (int j = sub; j < L; j += 16) {
      float pv = __expf(S[row][j] - m);
      s += pv;
      P[row][j] = (_Float16)pv;   // unnormalized; 1/sum folded into PV epilogue
    }
#pragma unroll
    for (int ofs = 8; ofs >= 1; ofs >>= 1) s += __shfl_xor(s, ofs, 32);
    if (sub == 0) rowsum[row] = s;
  }
  __syncthreads();

  // O = P @ V^T : wave -> (dk tile = wave&3, K half = wave>>2)
  int ntile = wave & 3, khalf = wave >> 2;
  int kb = khalf * (L >> 1);
  v8f acc = zero_v8f();
  for (int ks = 0; ks < (L >> 6); ++ks) {   // (L/2)/32 WMMA K-steps
    int kk = kb + ks * 32;
    v16h a = frag_A_rm(&P[0][kk], 512);
    v16h bf = frag_B_tr(vh + (size_t)(ntile * 16) * L + kk, L);
    acc = wmma_f16(a, bf, acc);
  }
  if (khalf == 1) {
#pragma unroll
    for (int r = 0; r < 8; ++r) redbuf[ntile][lane][r] = acc[r];
  }
  __syncthreads();
  if (khalf == 0) {
    int col = ntile * 16 + (lane & 15);
    int mh = (lane >> 4) * 8;
#pragma unroll
    for (int r = 0; r < 8; ++r) {
      float val = (acc[r] + redbuf[ntile][lane][r]) / rowsum[mh + r];
      size_t l = (size_t)blockIdx.x * 16 + mh + r;
      o[(((size_t)b * 4 + e) * L + l) * 512 + h * 64 + col] = (_Float16)val;
    }
  }
}

// ---------------------------------------------------------------------------
// K6: output projection + gate-weighted expert mix, double-buffered async.
// out[tok,n] = sum_e gate[b,e] * (o_e[tok,:] @ Wo[e]^T + bo[e])[n]
// grid: x = 512/64, y = Ntok/128.
// ---------------------------------------------------------------------------
__global__ __launch_bounds__(256) void outproj_kernel(
    const _Float16* __restrict__ ob, const _Float16* __restrict__ Woh,
    const float* __restrict__ bo, const float* __restrict__ gates,
    float* __restrict__ out, int L, int llog, int g) {
  __shared__ alignas(16) _Float16 At[2][128][48];
  __shared__ alignas(16) _Float16 Bs[2][64][40];
  int tid = threadIdx.x, wave = tid >> 5, lane = tid & 31;
  int wm = wave >> 1, wn = wave & 1;
  int mbase = blockIdx.y * 128, nbase = blockIdx.x * 64;
  int bidx = mbase >> llog;        // 128-token tile never crosses a batch row
  int lbase = mbase & (L - 1);
  float fin[2][2][8] = {};
  int mh = (lane >> 4) * 8;
  for (int e = 0; e < 4; ++e) {
    float gate = gates[(bidx * 3 + g) * 4 + e];
    const _Float16* Wh = Woh + (size_t)e * 512 * 512;
    const _Float16* osrc = ob + ((size_t)bidx * 4 + e) * (size_t)L * 512;

    auto stage = [&](int buf, int k0) {
#pragma unroll
      for (int c = 0; c < 2; ++c) {
        int idx = tid + c * 256;
        int m = idx >> 2, kk = (idx & 3) * 8;
        async_copy_b128(lds_off(&At[buf][m][kk]),
                        osrc + (size_t)(lbase + m) * 512 + k0 + kk);
      }
      {
        int n = tid >> 2, kk = (tid & 3) * 8;
        async_copy_b128(lds_off(&Bs[buf][n][kk]),
                        Wh + (size_t)(nbase + n) * 512 + k0 + kk);
      }
    };

    v8f acc[2][2];
#pragma unroll
    for (int i = 0; i < 2; ++i)
#pragma unroll
      for (int j = 0; j < 2; ++j) acc[i][j] = zero_v8f();

    __syncthreads();           // previous expert done reading LDS
    stage(0, 0);
    async_wait();
    __syncthreads();
    for (int s = 0; s < 16; ++s) {
      int cur = s & 1;
      if (s < 15) stage(cur ^ 1, (s + 1) * 32);
      v16h a0 = frag_A_rm(&At[cur][wm * 32][0], 48);
      v16h a1 = frag_A_rm(&At[cur][wm * 32 + 16][0], 48);
      v16h b0 = frag_B_tr(&Bs[cur][wn * 32][0], 40);
      v16h b1 = frag_B_tr(&Bs[cur][wn * 32 + 16][0], 40);
      acc[0][0] = wmma_f16(a0, b0, acc[0][0]);
      acc[0][1] = wmma_f16(a0, b1, acc[0][1]);
      acc[1][0] = wmma_f16(a1, b0, acc[1][0]);
      acc[1][1] = wmma_f16(a1, b1, acc[1][1]);
      if (s < 15) { async_wait(); __syncthreads(); }
    }
#pragma unroll
    for (int i = 0; i < 2; ++i)
#pragma unroll
      for (int j = 0; j < 2; ++j) {
        int ncol = nbase + wn * 32 + j * 16 + (lane & 15);
        float bb = bo[e * 512 + ncol];
#pragma unroll
        for (int r = 0; r < 8; ++r) fin[i][j][r] += gate * (acc[i][j][r] + bb);
      }
  }
#pragma unroll
  for (int i = 0; i < 2; ++i)
#pragma unroll
    for (int j = 0; j < 2; ++j) {
      int ncol = nbase + wn * 32 + j * 16 + (lane & 15);
#pragma unroll
      for (int r = 0; r < 8; ++r) {
        int m = mbase + wm * 32 + i * 16 + mh + r;
        out[(size_t)m * 512 + ncol] = fin[i][j][r];
      }
    }
}

// ---------------------------------------------------------------------------
// Host launcher
// ---------------------------------------------------------------------------
extern "C" void kernel_launch(void* const* d_in, const int* in_sizes, int n_in,
                              void* d_out, int out_size, void* d_ws, size_t ws_size,
                              hipStream_t stream) {
  (void)in_sizes; (void)n_in; (void)out_size; (void)ws_size;
  const float* x[3]   = {(const float*)d_in[0], (const float*)d_in[1], (const float*)d_in[2]};
  const float* gln_g  = (const float*)d_in[3];
  const float* gln_b  = (const float*)d_in[4];
  const float* emb    = (const float*)d_in[5];
  const float* emb_g  = (const float*)d_in[6];
  const float* emb_b  = (const float*)d_in[7];
  const float* gscale = (const float*)d_in[8];
  const float* mlp_g  = (const float*)d_in[9];
  const float* mlp_b  = (const float*)d_in[10];
  const float* W1 = (const float*)d_in[11]; const float* b1 = (const float*)d_in[12];
  const float* W2 = (const float*)d_in[13]; const float* b2 = (const float*)d_in[14];
  const float* Wg = (const float*)d_in[15]; const float* bg = (const float*)d_in[16];
  const float* Wq = (const float*)d_in[17]; const float* bq = (const float*)d_in[18];
  const float* Wk = (const float*)d_in[19]; const float* bk = (const float*)d_in[20];
  const float* Wv = (const float*)d_in[21]; const float* bv = (const float*)d_in[22];
  const float* Wo = (const float*)d_in[23]; const float* bo = (const float*)d_in[24];

  const int B = 8, D = 512, E = 4;
  const int Ls[3] = {512, 256, 128};
  const int Ll[3] = {9, 8, 7};

  // workspace carve-out (256B aligned)
  char* ws = (char*)d_ws;
  size_t off = 0;
  auto carve = [&](size_t bytes) -> char* {
    char* p = ws + off;
    off = (off + bytes + 255) & ~(size_t)255;
    return p;
  };
  _Float16* xn[3]; _Float16 *qb[3], *kb[3], *vb[3], *obuf[3];
  for (int g = 0; g < 3; ++g) xn[g] = (_Float16*)carve((size_t)B * Ls[g] * D * 2);
  float* pooled = (float*)carve((size_t)3 * B * D * 4);
  for (int g = 0; g < 3; ++g) {
    size_t sz = (size_t)B * E * Ls[g] * D * 2;
    qb[g]   = (_Float16*)carve(sz);
    kb[g]   = (_Float16*)carve(sz);
    vb[g]   = (_Float16*)carve(sz);   // stored transposed [b,e,h,dk,L]
    obuf[g] = (_Float16*)carve(sz);
  }
  const size_t wsz = (size_t)E * D * D;          // one weight stack, elems
  _Float16* Wqh = (_Float16*)carve(wsz * 2);
  _Float16* Wkh = (_Float16*)carve(wsz * 2);
  _Float16* Wvh = (_Float16*)carve(wsz * 2);
  _Float16* Woh = (_Float16*)carve(wsz * 2);

  // output carve-out
  float* out_g[3];
  out_g[0] = (float*)d_out;
  out_g[1] = out_g[0] + (size_t)B * 512 * D;
  out_g[2] = out_g[1] + (size_t)B * 256 * D;
  float* lat = out_g[2] + (size_t)B * 128 * D;
  float* gat = lat + (size_t)B * 3 * D;

  // one-shot fp32 -> f16 weight conversion (async-stageable operands)
  {
    int n8 = (int)(wsz / 8), blocks = (n8 + 255) / 256;
    wcvt_kernel<<<blocks, 256, 0, stream>>>(Wq, Wqh, n8);
    wcvt_kernel<<<blocks, 256, 0, stream>>>(Wk, Wkh, n8);
    wcvt_kernel<<<blocks, 256, 0, stream>>>(Wv, Wvh, n8);
    wcvt_kernel<<<blocks, 256, 0, stream>>>(Wo, Woh, n8);
  }

  for (int g = 0; g < 3; ++g) {
    int L = Ls[g];
    ln_kernel<<<B * L, 256, 0, stream>>>(x[g], gln_g + g * D, gln_b + g * D, xn[g]);
    pool_kernel<<<B, 256, 0, stream>>>(xn[g], pooled + (size_t)g * B * D, L);
  }
  router_kernel<<<1, 256, 0, stream>>>(pooled, emb, emb_g, emb_b, gscale,
                                       mlp_g, mlp_b, W1, b1, W2, b2, Wg, bg,
                                       lat, gat);
  for (int g = 0; g < 3; ++g) {
    int L = Ls[g], Ntok = B * L;
    proj_kernel<<<dim3(D / 64, Ntok / 128, 12), 256, 0, stream>>>(
        xn[g], Wqh, bq, Wkh, bk, Wvh, bv, qb[g], kb[g], vb[g], L, Ll[g]);
  }
  for (int g = 0; g < 3; ++g) {
    int L = Ls[g];
    attn_kernel<<<dim3(L / 16, B, E * 8), 256, 0, stream>>>(
        qb[g], kb[g], vb[g], obuf[g], L, 0.125f);
  }
  for (int g = 0; g < 3; ++g) {
    int L = Ls[g], Ntok = B * L;
    outproj_kernel<<<dim3(D / 64, Ntok / 128), 256, 0, stream>>>(
        obuf[g], Woh, bo, gat, out_g[g], L, Ll[g], g);
  }
}
